// UMABlock_74328704024963
// MI455X (gfx1250) — compile-verified, matrix operand order checked by reading, not run on hardware
//
#include <hip/hip_runtime.h>

// =====================================================================
// UMA-style equivariant GNN forward for MI455X (gfx1250, wave32, WMMA).
// All dense GEMMs (SO2 blocks, deg/rad/gate/FF MLPs) run through a
// bf16 WMMA tiled GEMM (v_wmma_f32_16x16x32_bf16, f32 accumulate) with
// double-buffered LDS tiles filled by GLOBAL_LOAD_ASYNC_TO_LDS_B128
// (ASYNCcnt-tracked, s_wait_asynccnt) so HBM latency overlaps WMMA.
// Weights are pre-transposed to [N][K] bf16 once, so tiles stage with
// pure b128 traffic. Edge-message tensors are kept in bf16 (halves HBM
// traffic vs f32 at 23.3 TB/s); node features / residuals stay f32;
// segment-sum uses global f32 atomics. Workspace requirement ~1.2 GB.
// =====================================================================

typedef unsigned short u16;
typedef unsigned int   u32;
typedef __attribute__((ext_vector_type(16))) __bf16 v16bf;
typedef __attribute__((ext_vector_type(8)))  float  v8f;

#define C_DIM 128
#define NSH   9
#define ROW   (NSH * C_DIM)          // 1152
#define NRBF  512
#define EIN   (NRBF + 2 * C_DIM)     // 768
#define CUT_F 5.0f
#define RESCALE_F 5.0f
#define PI_F  3.14159265358979f

#define BM 128
#define BN 128
#define BK 32

// ---------------------------------------------------------------------
// helpers
// ---------------------------------------------------------------------
__device__ __forceinline__ float bf2f(u16 h) {
  union { u32 u; float f; } x; x.u = ((u32)h) << 16; return x.f;
}
__device__ __forceinline__ u16 f2bf(float f) {
  union { float f; u32 u; } x; x.f = f;
  u32 r = x.u + 0x7FFFu + ((x.u >> 16) & 1u);
  return (u16)(r >> 16);
}
__device__ __forceinline__ float siluf(float x) { return x / (1.f + __expf(-x)); }
__device__ __forceinline__ float sigmf(float x) { return 1.f / (1.f + __expf(-x)); }

// CDNA5 async copy: global -> LDS, 16B per lane, tracked by ASYNCcnt.
__device__ __forceinline__ void async_b128(u32 lds_off, const void* gaddr) {
  asm volatile("global_load_async_to_lds_b128 %0, %1, off"
               :: "v"(lds_off), "v"(gaddr) : "memory");
}
__device__ __forceinline__ void wait_async0() {
  asm volatile("s_wait_asynccnt 0x0" ::: "memory");
}

__constant__ float BASIS[5][3][3] = {
  {{0.f, 0.7071067811865476f, 0.f}, {0.7071067811865476f, 0.f, 0.f}, {0.f, 0.f, 0.f}},
  {{0.f, 0.f, 0.f}, {0.f, 0.f, 0.7071067811865476f}, {0.f, 0.7071067811865476f, 0.f}},
  {{-0.4082482904638630f, 0.f, 0.f}, {0.f, -0.4082482904638630f, 0.f}, {0.f, 0.f, 0.8164965809277260f}},
  {{0.f, 0.f, 0.7071067811865476f}, {0.f, 0.f, 0.f}, {0.7071067811865476f, 0.f, 0.f}},
  {{0.7071067811865476f, 0.f, 0.f}, {0.f, -0.7071067811865476f, 0.f}, {0.f, 0.f, 0.f}}};

// ---------------------------------------------------------------------
// WMMA bf16 GEMM: C[M,N] = act(A[M,K] @ W[K,N] + bias)
// A bf16 row-major [M,K] (lda); B = W^T bf16 row-major [N,K] (ldb = K).
// 256 thr = 8 waves as 4x2; wave computes 32x64 via 2x4 WMMA 16x16x32.
// Double-buffered LDS filled with async b128 copies; one barrier/tile.
// ---------------------------------------------------------------------
__device__ __forceinline__ v16bf frag_ld(const u16* base) {
  // A/B fragment per CDNA5 ISA 7.12.2 (16-bit 16x32):
  // lane<16:  row=lane,     k = 0..7,16..23
  // lane>=16: row=lane-16,  k = 8..15,24..31
  int lane = threadIdx.x & 31;
  int r    = lane & 15;
  int kofs = (lane >> 4) << 3;
  union { uint4 q[2]; v16bf v; } u;
  const u16* p = base + r * BK + kofs;
  u.q[0] = *(const uint4*)(p);
  u.q[1] = *(const uint4*)(p + 16);
  return u.v;
}

__global__ void __launch_bounds__(256)
k_gemm(const u16* __restrict__ A, int lda,
       const u16* __restrict__ B, int ldb,   // B transposed: [N][K]
       const float* __restrict__ bias,
       void* __restrict__ Cout, int ldc,
       int M, int N, int K, int act, int outBf16)
{
  __shared__ __align__(16) u16 sA[2][BM][BK];
  __shared__ __align__(16) u16 sB[2][BN][BK];
  int tid  = threadIdx.x;
  int lane = tid & 31, wave = tid >> 5;
  int rowBlk = blockIdx.x * BM, colBlk = blockIdx.y * BN;
  int wm = (wave >> 1) * 32;   // 0,32,64,96
  int wn = (wave & 1) * 64;    // 0,64

  v8f acc[2][4] = {};

  int lrow = tid >> 2;         // 0..63
  int kseg = (tid & 3) * 8;

  // issue async copies for one 128x32 A tile + 128x32 B tile into buf
  auto issue = [&](int k0, int buf) {
#pragma unroll
    for (int i = 0; i < 2; i++) {
      int r = lrow + i * 64;
      if (rowBlk + r < M) {
        const u16* ap = A + (size_t)(rowBlk + r) * lda + k0 + kseg;
        async_b128((u32)(uintptr_t)&sA[buf][r][kseg], ap);
        if (k0 + BK < K) __builtin_prefetch(ap + BK, 0, 2);  // global_prefetch_b8
      } else {
        uint4 z = {0u, 0u, 0u, 0u};
        *(uint4*)(&sA[buf][r][kseg]) = z;   // zero-fill OOB rows (rare path)
      }
      async_b128((u32)(uintptr_t)&sB[buf][r][kseg],
                 B + (size_t)(colBlk + r) * ldb + k0 + kseg);
    }
  };

  const int ntiles = K / BK;
  issue(0, 0);

  for (int t = 0; t < ntiles; t++) {
    wait_async0();         // this wave's async writes for tile t are in LDS
    __syncthreads();       // ...and every other wave's too
    if (t + 1 < ntiles) issue((t + 1) * BK, (t + 1) & 1);

    const u16 (*cA)[BK] = sA[t & 1];
    const u16 (*cB)[BK] = sB[t & 1];
    v16bf af[2], bf[4];
#pragma unroll
    for (int im = 0; im < 2; im++) af[im] = frag_ld(&cA[wm + im * 16][0]);
#pragma unroll
    for (int in = 0; in < 4; in++) bf[in] = frag_ld(&cB[wn + in * 16][0]);
#pragma unroll
    for (int im = 0; im < 2; im++)
#pragma unroll
      for (int in = 0; in < 4; in++)
        acc[im][in] = __builtin_amdgcn_wmma_f32_16x16x32_bf16(
            false, af[im], false, bf[in], (short)0, acc[im][in], false, false);
  }

  // epilogue: D layout — VGPR r: lanes 0-15 M=r, lanes 16-31 M=r+8; N=lane&15
  int ro = (lane >> 4) << 3;
  int co = lane & 15;
#pragma unroll
  for (int im = 0; im < 2; im++)
#pragma unroll
    for (int in = 0; in < 4; in++)
#pragma unroll
      for (int r = 0; r < 8; r++) {
        int row = rowBlk + wm + im * 16 + r + ro;
        int col = colBlk + wn + in * 16 + co;
        if (row < M) {
          float v = acc[im][in][r];
          if (bias) v += bias[col];
          if (act == 1) v = siluf(v);
          else if (act == 2) v = sigmf(v);
          if (outBf16) ((u16*)Cout)[(size_t)row * ldc + col] = f2bf(v);
          else         ((float*)Cout)[(size_t)row * ldc + col] = v;
        }
      }
}

// ---------------------------------------------------------------------
// weight prep: convert f32 [K,N] -> bf16 transposed [N,K]
// ---------------------------------------------------------------------
__global__ void k_cvt_t(const float* __restrict__ s, u16* __restrict__ d, int K, int N) {
  int i = blockIdx.x * blockDim.x + threadIdx.x;
  if (i >= K * N) return;
  int k = i / N, n = i % N;
  d[(size_t)n * K + k] = f2bf(s[i]);
}

// fused complex block [[wr, wi], [-wi, wr]] from wr,wi [K0,N0], stored
// transposed: dst[(2N0)][(2K0)]
__global__ void k_build_rot_t(const float* __restrict__ wr, const float* __restrict__ wi,
                              u16* __restrict__ dst, int K0, int N0) {
  int i = blockIdx.x * blockDim.x + threadIdx.x;
  int tot = 4 * K0 * N0;
  if (i >= tot) return;
  int n2w = 2 * N0;
  int k2 = i / n2w, n2 = i % n2w;
  float v;
  if (k2 < K0) v = (n2 < N0) ? wr[k2 * N0 + n2] : wi[k2 * N0 + n2 - N0];
  else         v = (n2 < N0) ? -wi[(k2 - K0) * N0 + n2] : wr[(k2 - K0) * N0 + n2 - N0];
  dst[(size_t)n2 * (2 * K0) + k2] = f2bf(v);
}

// ---------------------------------------------------------------------
// per-edge geometry: Wigner 9x9 (natural order), distance, poly cutoff
// ---------------------------------------------------------------------
__global__ void k_edge_geom(const float* __restrict__ ev, const float* __restrict__ gamma,
                            float* __restrict__ W, float* __restrict__ dist,
                            float* __restrict__ env, int E)
{
  int e = blockIdx.x * blockDim.x + threadIdx.x;
  if (e >= E) return;
  float x = ev[3 * e], y = ev[3 * e + 1], z = ev[3 * e + 2];
  float d = sqrtf(x * x + y * y + z * z + 1e-12f);
  float id = 1.f / d;
  float ux = x * id, uy = y * id, uz = z * id;
  float vx = uy, vy = -ux;
  float denom = fmaxf(1.f + uz, 1e-6f);
  float K[3][3] = {{0.f, 0.f, vy}, {0.f, 0.f, -vx}, {-vy, vx, 0.f}};
  float K2[3][3];
#pragma unroll
  for (int i = 0; i < 3; i++)
#pragma unroll
    for (int j = 0; j < 3; j++) {
      float s = 0.f;
#pragma unroll
      for (int k = 0; k < 3; k++) s += K[i][k] * K[k][j];
      K2[i][j] = s;
    }
  float R0[3][3];
#pragma unroll
  for (int i = 0; i < 3; i++)
#pragma unroll
    for (int j = 0; j < 3; j++)
      R0[i][j] = (i == j ? 1.f : 0.f) + K[i][j] + K2[i][j] / denom;
  float g = gamma[e];
  float cg = __cosf(g), sg = __sinf(g);
  float Rz[3][3] = {{cg, -sg, 0.f}, {sg, cg, 0.f}, {0.f, 0.f, 1.f}};
  float R[3][3];
#pragma unroll
  for (int i = 0; i < 3; i++)
#pragma unroll
    for (int j = 0; j < 3; j++) {
      float s = 0.f;
#pragma unroll
      for (int k = 0; k < 3; k++) s += Rz[i][k] * R0[k][j];
      R[i][j] = s;
    }
  // D1 = P R P^T with P = cyclic perm (y,z,x)
  const int a[3] = {1, 2, 0};
  float D1[3][3];
#pragma unroll
  for (int m = 0; m < 3; m++)
#pragma unroll
    for (int n = 0; n < 3; n++) D1[m][n] = R[a[m]][a[n]];
  // D2[m][n] = <B_m, R B_n R^T>
  float D2[5][5];
  float RB[5][3][3];
#pragma unroll
  for (int n = 0; n < 5; n++) {
    float T[3][3];
#pragma unroll
    for (int i = 0; i < 3; i++)
#pragma unroll
      for (int k = 0; k < 3; k++) {
        float s = 0.f;
#pragma unroll
        for (int j = 0; j < 3; j++) s += R[i][j] * BASIS[n][j][k];
        T[i][k] = s;
      }
#pragma unroll
    for (int i = 0; i < 3; i++)
#pragma unroll
      for (int l = 0; l < 3; l++) {
        float s = 0.f;
#pragma unroll
        for (int k = 0; k < 3; k++) s += T[i][k] * R[l][k];
        RB[n][i][l] = s;
      }
  }
#pragma unroll
  for (int m = 0; m < 5; m++)
#pragma unroll
    for (int n = 0; n < 5; n++) {
      float s = 0.f;
#pragma unroll
      for (int i = 0; i < 3; i++)
#pragma unroll
        for (int l = 0; l < 3; l++) s += BASIS[m][i][l] * RB[n][i][l];
      D2[m][n] = s;
    }
  float* We = W + (size_t)e * 81;
#pragma unroll
  for (int t = 0; t < 81; t++) We[t] = 0.f;
  We[0] = 1.f;
#pragma unroll
  for (int m = 0; m < 3; m++)
#pragma unroll
    for (int n = 0; n < 3; n++) We[(1 + m) * 9 + (1 + n)] = D1[m][n];
#pragma unroll
  for (int m = 0; m < 5; m++)
#pragma unroll
    for (int n = 0; n < 5; n++) We[(4 + m) * 9 + (4 + n)] = D2[m][n];
  dist[e] = d;
  float xr = d / CUT_F;
  float x2 = xr * xr, x3 = x2 * xr, x6 = x3 * x3, x7 = x6 * xr, x8 = x7 * xr;
  float f = 1.f - 28.f * x6 + 48.f * x7 - 21.f * x8;
  env[e] = (xr < 1.f) ? f : 0.f;
}

// ---------------------------------------------------------------------
// edge embeddings [E, 768] bf16 : gaussian rbf | send_emb | recv_emb
// ---------------------------------------------------------------------
__global__ void k_edge_embed(const float* __restrict__ dist, const int* __restrict__ species,
                             const int* __restrict__ snd, const int* __restrict__ rcv,
                             const float* __restrict__ semb, const float* __restrict__ remb,
                             u16* __restrict__ out, int E)
{
  long i = (long)blockIdx.x * blockDim.x + threadIdx.x;
  long tot = (long)E * EIN;
  if (i >= tot) return;
  int e = (int)(i / EIN), k = (int)(i % EIN);
  float v;
  if (k < NRBF) {
    float off = CUT_F * (float)k / (float)(NRBF - 1);
    float w = 2.f * CUT_F / (float)(NRBF - 1);
    float t = (dist[e] - off) / w;
    v = __expf(-0.5f * t * t);
  } else if (k < NRBF + C_DIM) {
    v = semb[species[snd[e]] * C_DIM + (k - NRBF)];
  } else {
    v = remb[species[rcv[e]] * C_DIM + (k - NRBF - C_DIM)];
  }
  out[i] = f2bf(v);
}

// ---------------------------------------------------------------------
// node init: row0 = sphere_emb[spec] + csd[graph], rows 1..8 = 0
// ---------------------------------------------------------------------
__global__ void k_node_init(const int* __restrict__ species, const float* __restrict__ csd,
                            const int* __restrict__ n_node, const float* __restrict__ semb,
                            float* __restrict__ node, int N, int G)
{
  long i = (long)blockIdx.x * blockDim.x + threadIdx.x;
  if (i >= (long)N * C_DIM) return;
  int n = (int)(i / C_DIM), c = (int)(i % C_DIM);
  int g = 0, acc = 0;
  for (int t = 0; t < G; t++) { acc += n_node[t]; if (n >= acc) g = t + 1; }
  if (g >= G) g = G - 1;
  node[(size_t)n * ROW + c] = semb[species[n] * C_DIM + c] + csd[g * C_DIM + c];
#pragma unroll
  for (int j = 1; j < NSH; j++) node[(size_t)n * ROW + j * C_DIM + c] = 0.f;
}

// ---------------------------------------------------------------------
// norm_sh: layernorm on l=0, rms norm on l=1 (rows 1..3), l=2 (rows 4..8)
// block = 128 threads (one node)
// ---------------------------------------------------------------------
__global__ void k_norm_sh(const float* __restrict__ x,
                          const float* __restrict__ w, const float* __restrict__ b,
                          const float* __restrict__ wl,
                          float* __restrict__ out, int N)
{
  int n = blockIdx.x, c = threadIdx.x;
  const float* xr = x + (size_t)n * ROW;
  float* orow = out + (size_t)n * ROW;
  __shared__ float red[C_DIM];

  float x0 = xr[c];
  red[c] = x0; __syncthreads();
  for (int s = 64; s > 0; s >>= 1) { if (c < s) red[c] += red[c + s]; __syncthreads(); }
  float mu = red[0] * (1.f / C_DIM); __syncthreads();
  float dx = x0 - mu;
  red[c] = dx * dx; __syncthreads();
  for (int s = 64; s > 0; s >>= 1) { if (c < s) red[c] += red[c + s]; __syncthreads(); }
  float var = red[0] * (1.f / C_DIM); __syncthreads();
  orow[c] = dx * rsqrtf(var + 1e-5f) * w[c] + b[c];

  float s1 = 0.f;
#pragma unroll
  for (int j = 1; j < 4; j++) { float t = xr[j * C_DIM + c]; s1 += t * t; }
  red[c] = s1; __syncthreads();
  for (int s = 64; s > 0; s >>= 1) { if (c < s) red[c] += red[c + s]; __syncthreads(); }
  float inv1 = rsqrtf(red[0] * (1.f / (3 * C_DIM)) + 1e-5f); __syncthreads();
#pragma unroll
  for (int j = 1; j < 4; j++) orow[j * C_DIM + c] = xr[j * C_DIM + c] * inv1 * wl[c];

  float s2 = 0.f;
#pragma unroll
  for (int j = 4; j < 9; j++) { float t = xr[j * C_DIM + c]; s2 += t * t; }
  red[c] = s2; __syncthreads();
  for (int s = 64; s > 0; s >>= 1) { if (c < s) red[c] += red[c + s]; __syncthreads(); }
  float inv2 = rsqrtf(red[0] * (1.f / (5 * C_DIM)) + 1e-5f);
#pragma unroll
  for (int j = 4; j < 9; j++) orow[j * C_DIM + c] = xr[j * C_DIM + c] * inv2 * wl[C_DIM + c];
}

// ---------------------------------------------------------------------
// gather msg = x[snd]+x[rcv], rotate by W, store bf16 in SO2 order
// SO2 slot order (natural sh idx): {0,2,6, 3,7, 1,5, 8, 4}
// ---------------------------------------------------------------------
__global__ void k_gather_rotate(const float* __restrict__ xn, const float* __restrict__ W,
                                const int* __restrict__ snd, const int* __restrict__ rcv,
                                u16* __restrict__ out, int E)
{
  int e = blockIdx.x, c = threadIdx.x;
  __shared__ float sW[81];
  if (c < 81) sW[c] = W[(size_t)e * 81 + c];
  __syncthreads();
  int s = snd[e], r = rcv[e];
  const float* xs = xn + (size_t)s * ROW + c;
  const float* xr = xn + (size_t)r * ROW + c;
  float m[9];
#pragma unroll
  for (int j = 0; j < 9; j++) m[j] = xs[j * C_DIM] + xr[j * C_DIM];
  const int ord[9] = {0, 2, 6, 3, 7, 1, 5, 8, 4};
  u16* o = out + (size_t)e * ROW + c;
#pragma unroll
  for (int t = 0; t < 9; t++) {
    int i = ord[t];
    float v = 0.f;
#pragma unroll
    for (int j = 0; j < 9; j++) v += sW[i * 9 + j] * m[j];
    o[t * C_DIM] = f2bf(v);
  }
}

// rotate back (W^T), * env, atomic segment-sum into node features
__global__ void k_rotate_back_scatter(const u16* __restrict__ msg, const float* __restrict__ W,
                                      const float* __restrict__ env, const int* __restrict__ rcv,
                                      float* __restrict__ node, int E)
{
  int e = blockIdx.x, c = threadIdx.x;
  __shared__ float sW[81];
  if (c < 81) sW[c] = W[(size_t)e * 81 + c];
  __syncthreads();
  const int ord[9] = {0, 2, 6, 3, 7, 1, 5, 8, 4};
  float mnat[9];
  const u16* mi = msg + (size_t)e * ROW + c;
#pragma unroll
  for (int t = 0; t < 9; t++) mnat[ord[t]] = bf2f(mi[t * C_DIM]);
  float ev = env[e];
  float* nd = node + (size_t)rcv[e] * ROW + c;
#pragma unroll
  for (int i = 0; i < 9; i++) {
    float v = 0.f;
#pragma unroll
    for (int j = 0; j < 9; j++) v += sW[j * 9 + i] * mnat[j];
    atomicAdd(nd + i * C_DIM, v * ev);
  }
}

// initial degree embedding: x_edge rows (0,2,6)=r, W^T rotate, *env/RESCALE
__global__ void k_xedge_scatter(const float* __restrict__ rdeg, const float* __restrict__ W,
                                const float* __restrict__ env, const int* __restrict__ rcv,
                                float* __restrict__ node, int E)
{
  int e = blockIdx.x, c = threadIdx.x;
  __shared__ float sW[81];
  if (c < 81) sW[c] = W[(size_t)e * 81 + c];
  __syncthreads();
  const float* rr = rdeg + (size_t)e * 384;
  float r0 = rr[c], r1 = rr[C_DIM + c], r2 = rr[2 * C_DIM + c];
  float sc = env[e] * (1.f / RESCALE_F);
  float* nd = node + (size_t)rcv[e] * ROW + c;
#pragma unroll
  for (int i = 0; i < 9; i++) {
    float v = (sW[0 * 9 + i] * r0 + sW[2 * 9 + i] * r1 + sW[6 * 9 + i] * r2) * sc;
    atomicAdd(nd + i * C_DIM, v);
  }
}

// msg *= rad (broadcast over the 9 SO2 slots), in-place bf16
__global__ void k_rad_mul(u16* __restrict__ buf, const float* __restrict__ rad, long total) {
  long i = (long)blockIdx.x * blockDim.x + threadIdx.x;
  if (i >= total) return;
  long e = i / ROW; int h = (int)(i % C_DIM);
  buf[i] = f2bf(bf2f(buf[i]) * rad[e * C_DIM + h]);
}

// silu on slot0; *g1 on l=1 slots {1,3,5}; *g2 on l=2 slots {2,4,6,7,8}
__global__ void k_gate_apply(u16* __restrict__ buf, const float* __restrict__ gates, long total) {
  long i = (long)blockIdx.x * blockDim.x + threadIdx.x;
  if (i >= total) return;
  long e = i / ROW;
  int t = (int)((i / C_DIM) % NSH);
  int h = (int)(i % C_DIM);
  float v = bf2f(buf[i]);
  if (t == 0) v = siluf(v);
  else {
    bool l1 = (t == 1) || (t == 3) || (t == 5);
    v *= gates[e * (2 * C_DIM) + (l1 ? h : C_DIM + h)];
  }
  buf[i] = f2bf(v);
}

// ---------------------------------------------------------------------
// grid transforms (TO_GRID [64,9], FROM_GRID [9,64] built on device)
// ---------------------------------------------------------------------
__global__ void k_make_grid(float* __restrict__ TG, float* __restrict__ FG) {
  if (threadIdx.x != 0 || blockIdx.x != 0) return;
  float Y[64][9];
  for (int ti = 0; ti < 8; ti++)
    for (int pj = 0; pj < 8; pj++) {
      float th = (ti + 0.5f) * (PI_F / 8.f);
      float ph = (pj + 0.5f) * (2.f * PI_F / 8.f);
      float st = __sinf(th), ct = __cosf(th), cp = __cosf(ph), sp = __sinf(ph);
      float x = st * cp, y = st * sp, z = ct;
      int p = ti * 8 + pj;
      Y[p][0] = 1.f; Y[p][1] = y; Y[p][2] = z; Y[p][3] = x;
      Y[p][4] = x * y; Y[p][5] = y * z; Y[p][6] = 3.f * z * z - 1.f;
      Y[p][7] = x * z; Y[p][8] = x * x - y * y;
    }
  float Gm[9][9], Gi[9][9];
  for (int i = 0; i < 9; i++)
    for (int j = 0; j < 9; j++) {
      float s = 0.f;
      for (int p = 0; p < 64; p++) s += Y[p][i] * Y[p][j];
      Gm[i][j] = s; Gi[i][j] = (i == j) ? 1.f : 0.f;
    }
  for (int k = 0; k < 9; k++) {
    float piv = 1.f / Gm[k][k];
    for (int j = 0; j < 9; j++) { Gm[k][j] *= piv; Gi[k][j] *= piv; }
    for (int i = 0; i < 9; i++)
      if (i != k) {
        float f = Gm[i][k];
        for (int j = 0; j < 9; j++) { Gm[i][j] -= f * Gm[k][j]; Gi[i][j] -= f * Gi[k][j]; }
      }
  }
  for (int p = 0; p < 64; p++)
    for (int l = 0; l < 9; l++) TG[p * 9 + l] = Y[p][l];
  for (int l = 0; l < 9; l++)
    for (int p = 0; p < 64; p++) {
      float s = 0.f;
      for (int m = 0; m < 9; m++) s += Gi[l][m] * Y[p][m];
      FG[l * 64 + p] = s;
    }
}

__global__ void __launch_bounds__(256)
k_to_grid(const float* __restrict__ x, const float* __restrict__ TG,
          u16* __restrict__ g, int N)
{
  int n = blockIdx.x, tid = threadIdx.x;
  __shared__ float sx[ROW];
  __shared__ float sT[64 * 9];
  for (int i = tid; i < ROW; i += 256) sx[i] = x[(size_t)n * ROW + i];
  for (int i = tid; i < 576; i += 256) sT[i] = TG[i];
  __syncthreads();
  for (int i = tid; i < 64 * C_DIM; i += 256) {
    int p = i >> 7, c = i & 127;
    float v = 0.f;
#pragma unroll
    for (int l = 0; l < 9; l++) v += sT[p * 9 + l] * sx[l * C_DIM + c];
    g[((size_t)n * 64 + p) * C_DIM + c] = f2bf(v);
  }
}

__global__ void k_from_grid(const u16* __restrict__ g, const float* __restrict__ FG,
                            float* __restrict__ node, int N)
{
  int n = blockIdx.x, c = threadIdx.x;
  __shared__ float sF[9 * 64];
  for (int i = c; i < 576; i += 128) sF[i] = FG[i];
  __syncthreads();
  float acc[9] = {0.f, 0.f, 0.f, 0.f, 0.f, 0.f, 0.f, 0.f, 0.f};
  for (int p = 0; p < 64; p++) {
    float gv = bf2f(g[((size_t)n * 64 + p) * C_DIM + c]);
#pragma unroll
    for (int l = 0; l < 9; l++) acc[l] += sF[l * 64 + p] * gv;
  }
#pragma unroll
  for (int l = 0; l < 9; l++) node[(size_t)n * ROW + l * C_DIM + c] += acc[l];  // residual in-place
}

// =====================================================================
// host orchestration
// =====================================================================
extern "C" void kernel_launch(void* const* d_in, const int* in_sizes, int n_in,
                              void* d_out, int out_size, void* d_ws, size_t ws_size,
                              hipStream_t stream)
{
  (void)n_in; (void)out_size;
  const float* edge_vectors = (const float*)d_in[0];
  const int*   species      = (const int*)d_in[1];
  const float* csd          = (const float*)d_in[2];
  const int*   senders      = (const int*)d_in[3];
  const int*   receivers    = (const int*)d_in[4];
  const int*   n_node       = (const int*)d_in[5];
  const float* rot_gamma    = (const float*)d_in[6];
  const int E = in_sizes[4];
  const int N = in_sizes[1];
  const int G = in_sizes[5];

  // params pytree, JAX flatten order = sorted dict keys, lists in order
  const float* deg_b[3] = {(const float*)d_in[7], (const float*)d_in[8], (const float*)d_in[9]};
  const float* deg_w[3] = {(const float*)d_in[10], (const float*)d_in[11], (const float*)d_in[12]};
  const float* fn_b  = (const float*)d_in[13];
  const float* fn_w  = (const float*)d_in[14];
  const float* fn_wl = (const float*)d_in[15];
  struct Lay {
    const float *ff_b[3], *ff_w[3], *gate_b, *gate_w;
    const float *n1_b, *n1_w, *n1_wl, *n2_b, *n2_w, *n2_wl;
    const float *rad_b[2], *rad_w[2];
    const float *s1_w0, *s1_w1i, *s1_w1r, *s1_w2i, *s1_w2r;
    const float *s2_w0, *s2_w1i, *s2_w1r, *s2_w2i, *s2_w2r;
  } L[2];
  for (int l = 0; l < 2; l++) {
    const int o = 16 + 28 * l;
    L[l].ff_b[0] = (const float*)d_in[o + 0];  L[l].ff_b[1] = (const float*)d_in[o + 1];
    L[l].ff_b[2] = (const float*)d_in[o + 2];
    L[l].ff_w[0] = (const float*)d_in[o + 3];  L[l].ff_w[1] = (const float*)d_in[o + 4];
    L[l].ff_w[2] = (const float*)d_in[o + 5];
    L[l].gate_b  = (const float*)d_in[o + 6];  L[l].gate_w  = (const float*)d_in[o + 7];
    L[l].n1_b = (const float*)d_in[o + 8];  L[l].n1_w = (const float*)d_in[o + 9];
    L[l].n1_wl = (const float*)d_in[o + 10];
    L[l].n2_b = (const float*)d_in[o + 11]; L[l].n2_w = (const float*)d_in[o + 12];
    L[l].n2_wl = (const float*)d_in[o + 13];
    L[l].rad_b[0] = (const float*)d_in[o + 14]; L[l].rad_b[1] = (const float*)d_in[o + 15];
    L[l].rad_w[0] = (const float*)d_in[o + 16]; L[l].rad_w[1] = (const float*)d_in[o + 17];
    L[l].s1_w0 = (const float*)d_in[o + 18]; L[l].s1_w1i = (const float*)d_in[o + 19];
    L[l].s1_w1r = (const float*)d_in[o + 20]; L[l].s1_w2i = (const float*)d_in[o + 21];
    L[l].s1_w2r = (const float*)d_in[o + 22];
    L[l].s2_w0 = (const float*)d_in[o + 23]; L[l].s2_w1i = (const float*)d_in[o + 24];
    L[l].s2_w1r = (const float*)d_in[o + 25]; L[l].s2_w2i = (const float*)d_in[o + 26];
    L[l].s2_w2r = (const float*)d_in[o + 27];
  }
  const float* recv_emb   = (const float*)d_in[72];
  const float* send_emb   = (const float*)d_in[73];
  const float* sphere_emb = (const float*)d_in[74];

  // ---------------- workspace layout ----------------
  char* wb = (char*)d_ws;
  size_t off = 0;
  auto alloc = [&](size_t bytes) -> void* {
    void* p = wb + off;
    off += (bytes + 255) & ~(size_t)255;
    return p;
  };
  float* TG   = (float*)alloc(64 * 9 * 4);
  float* FG   = (float*)alloc(9 * 64 * 4);
  float* Wg   = (float*)alloc((size_t)E * 81 * 4);
  float* dist = (float*)alloc((size_t)E * 4);
  float* env  = (float*)alloc((size_t)E * 4);
  u16*   eemb = (u16*)alloc((size_t)E * EIN * 2);
  float* node = (float*)alloc((size_t)N * ROW * 4);
  float* xnrm = (float*)alloc((size_t)N * ROW * 4);
  u16*   Abuf = (u16*)alloc((size_t)E * ROW * 2);   // rotated msg / grid ping / deg-out alias
  u16*   Bbuf = (u16*)alloc((size_t)E * ROW * 2);   // so2_2 output
  u16*   Cbuf = (u16*)alloc((size_t)E * ROW * 2);   // so2_1 output / grid pong
  float* radb  = (float*)alloc((size_t)E * C_DIM * 4);
  float* gates = (float*)alloc((size_t)E * 2 * C_DIM * 4);
  u16* t1 = (u16*)alloc((size_t)E * C_DIM * 2);
  u16* t2 = (u16*)alloc((size_t)E * C_DIM * 2);
  // bf16 weights (transposed [N][K])
  u16* degW0b = (u16*)alloc(EIN * C_DIM * 2);
  u16* degW1b = (u16*)alloc(C_DIM * C_DIM * 2);
  u16* degW2b = (u16*)alloc(C_DIM * 384 * 2);
  u16 *s1w0b[2], *s1w1b[2], *s1w2b[2], *s2w0b[2], *s2w1b[2], *s2w2b[2];
  u16 *radW0b[2], *radW1b[2], *gateWb[2], *ffWb[2][3];
  for (int l = 0; l < 2; l++) {
    s1w0b[l] = (u16*)alloc(384 * 384 * 2);
    s1w1b[l] = (u16*)alloc(512 * 512 * 2);
    s1w2b[l] = (u16*)alloc(256 * 256 * 2);
    s2w0b[l] = (u16*)alloc(384 * 384 * 2);
    s2w1b[l] = (u16*)alloc(512 * 512 * 2);
    s2w2b[l] = (u16*)alloc(256 * 256 * 2);
    radW0b[l] = (u16*)alloc(EIN * C_DIM * 2);
    radW1b[l] = (u16*)alloc(C_DIM * C_DIM * 2);
    gateWb[l] = (u16*)alloc(C_DIM * 256 * 2);
    for (int m = 0; m < 3; m++) ffWb[l][m] = (u16*)alloc(C_DIM * C_DIM * 2);
  }
  float* rdeg = (float*)(void*)Abuf;  // [E,384] f32 alias, only before layer loop
  if (off > ws_size) return;          // workspace too small; nothing safe to do

  auto cvt = [&](const float* s, u16* d, int K, int Nn) {
    int n = K * Nn;
    k_cvt_t<<<(n + 255) / 256, 256, 0, stream>>>(s, d, K, Nn);
  };
  auto bld = [&](const float* wr, const float* wi, u16* d, int K0, int N0) {
    int n = 4 * K0 * N0;
    k_build_rot_t<<<(n + 255) / 256, 256, 0, stream>>>(wr, wi, d, K0, N0);
  };
  auto gemm = [&](const u16* A, int lda, const u16* B, const float* bias,
                  void* Cp, int ldc, int M, int Nn, int K, int act, int obf) {
    dim3 g((M + BM - 1) / BM, Nn / BN);
    k_gemm<<<g, 256, 0, stream>>>(A, lda, B, K, bias, Cp, ldc, M, Nn, K, act, obf);
  };

  // ---------------- setup ----------------
  k_make_grid<<<1, 1, 0, stream>>>(TG, FG);
  k_edge_geom<<<(E + 255) / 256, 256, 0, stream>>>(edge_vectors, rot_gamma, Wg, dist, env, E);
  {
    long tot = (long)E * EIN;
    k_edge_embed<<<(unsigned)((tot + 255) / 256), 256, 0, stream>>>(
        dist, species, senders, receivers, send_emb, recv_emb, eemb, E);
  }
  {
    long tot = (long)N * C_DIM;
    k_node_init<<<(unsigned)((tot + 255) / 256), 256, 0, stream>>>(
        species, csd, n_node, sphere_emb, node, N, G);
  }
  cvt(deg_w[0], degW0b, EIN, C_DIM);
  cvt(deg_w[1], degW1b, C_DIM, C_DIM);
  cvt(deg_w[2], degW2b, C_DIM, 384);
  for (int l = 0; l < 2; l++) {
    cvt(L[l].s1_w0, s1w0b[l], 384, 384);
    bld(L[l].s1_w1r, L[l].s1_w1i, s1w1b[l], 256, 256);
    bld(L[l].s1_w2r, L[l].s1_w2i, s1w2b[l], 128, 128);
    cvt(L[l].s2_w0, s2w0b[l], 384, 384);
    bld(L[l].s2_w1r, L[l].s2_w1i, s2w1b[l], 256, 256);
    bld(L[l].s2_w2r, L[l].s2_w2i, s2w2b[l], 128, 128);
    cvt(L[l].rad_w[0], radW0b[l], EIN, C_DIM);
    cvt(L[l].rad_w[1], radW1b[l], C_DIM, C_DIM);
    cvt(L[l].gate_w, gateWb[l], C_DIM, 256);
    for (int m = 0; m < 3; m++) cvt(L[l].ff_w[m], ffWb[l][m], C_DIM, C_DIM);
  }

  // ---------------- degree embedding ----------------
  gemm(eemb, EIN, degW0b, deg_b[0], t1, C_DIM, E, C_DIM, EIN, 1, 1);
  gemm(t1, C_DIM, degW1b, deg_b[1], t2, C_DIM, E, C_DIM, C_DIM, 1, 1);
  gemm(t2, C_DIM, degW2b, deg_b[2], rdeg, 384, E, 384, C_DIM, 0, 0);
  k_xedge_scatter<<<E, 128, 0, stream>>>(rdeg, Wg, env, receivers, node, E);

  // ---------------- layers ----------------
  for (int l = 0; l < 2; l++) {
    k_norm_sh<<<N, 128, 0, stream>>>(node, L[l].n1_w, L[l].n1_b, L[l].n1_wl, xnrm, N);
    k_gather_rotate<<<E, 128, 0, stream>>>(xnrm, Wg, senders, receivers, Abuf, E);
    // so2_1 (ci=C -> co=H), SO2-ordered strided GEMMs
    gemm(Abuf + 0,   ROW, s1w0b[l], nullptr, Cbuf + 0,   ROW, E, 384, 384, 0, 1);
    gemm(Abuf + 384, ROW, s1w1b[l], nullptr, Cbuf + 384, ROW, E, 512, 512, 0, 1);
    gemm(Abuf + 896, ROW, s1w2b[l], nullptr, Cbuf + 896, ROW, E, 256, 256, 0, 1);
    // radial MLP
    gemm(eemb, EIN, radW0b[l], L[l].rad_b[0], t1, C_DIM, E, C_DIM, EIN, 1, 1);
    gemm(t1, C_DIM, radW1b[l], L[l].rad_b[1], radb, C_DIM, E, C_DIM, C_DIM, 0, 0);
    {
      long tot = (long)E * ROW;
      k_rad_mul<<<(unsigned)((tot + 255) / 256), 256, 0, stream>>>(Cbuf, radb, tot);
    }
    // gates from slot0 (natural row 0)
    gemm(Cbuf + 0, ROW, gateWb[l], L[l].gate_b, gates, 256, E, 256, C_DIM, 2, 0);
    {
      long tot = (long)E * ROW;
      k_gate_apply<<<(unsigned)((tot + 255) / 256), 256, 0, stream>>>(Cbuf, gates, tot);
    }
    // so2_2 (ci=H -> co=C)
    gemm(Cbuf + 0,   ROW, s2w0b[l], nullptr, Bbuf + 0,   ROW, E, 384, 384, 0, 1);
    gemm(Cbuf + 384, ROW, s2w1b[l], nullptr, Bbuf + 384, ROW, E, 512, 512, 0, 1);
    gemm(Cbuf + 896, ROW, s2w2b[l], nullptr, Bbuf + 896, ROW, E, 256, 256, 0, 1);
    k_rotate_back_scatter<<<E, 128, 0, stream>>>(Bbuf, Wg, env, receivers, node, E);
    // grid FF (residual added in k_from_grid)
    k_norm_sh<<<N, 128, 0, stream>>>(node, L[l].n2_w, L[l].n2_b, L[l].n2_wl, xnrm, N);
    k_to_grid<<<N, 256, 0, stream>>>(xnrm, TG, Abuf, N);
    gemm(Abuf, C_DIM, ffWb[l][0], L[l].ff_b[0], Cbuf, C_DIM, N * 64, C_DIM, C_DIM, 1, 1);
    gemm(Cbuf, C_DIM, ffWb[l][1], L[l].ff_b[1], Abuf, C_DIM, N * 64, C_DIM, C_DIM, 1, 1);
    gemm(Abuf, C_DIM, ffWb[l][2], L[l].ff_b[2], Cbuf, C_DIM, N * 64, C_DIM, C_DIM, 0, 1);
    k_from_grid<<<N, 128, 0, stream>>>(Cbuf, FG, node, N);
  }

  // ---------------- final norm -> output [N, 9, 128] f32 ----------------
  k_norm_sh<<<N, 128, 0, stream>>>(node, fn_w, fn_b, fn_wl, (float*)d_out, N);
}
